// ContrastLoss_34720515621627
// MI455X (gfx1250) — matrix operand order repaired and checked by reading, not compiled, not run on hardware
//
#include <hip/hip_runtime.h>
#include <hip/hip_bf16.h>
#include <math.h>

#define NROWS 8192
#define HID 64
#define INV_TAU 1.25f     // 1/0.8
#define LAMBDA 0.5f
#define NTILES 64         // NROWS / 128

typedef __attribute__((ext_vector_type(16))) _Float16 v16h;
typedef __attribute__((ext_vector_type(8)))  _Float16 v8h;
typedef __attribute__((ext_vector_type(8)))  float    v8f;

// ---------------------------------------------------------------------------
// 1) pack positive_matrix (f32 binary, 256MB) into a bitmask (8MB, L2-resident)
//    one wave32 handles 32 consecutive elements -> one u32 word via ballot
// ---------------------------------------------------------------------------
__global__ __launch_bounds__(256) void cl_pack_kernel(const float* __restrict__ P,
                                                      unsigned* __restrict__ Pb) {
    long long gid = (long long)blockIdx.x * 256 + threadIdx.x;   // [0, 64M)
    float v = __builtin_nontemporal_load(&P[gid]);
    unsigned long long m = __ballot(v != 0.0f);   // wave32: bits [31:0]
    if ((threadIdx.x & 31) == 0) Pb[gid >> 5] = (unsigned)m;
}

// ---------------------------------------------------------------------------
// 2) projection: z = (ELU(x@W1^T + b1))@W2^T + b2 ; L2-normalize ; cast f16
// ---------------------------------------------------------------------------
__global__ __launch_bounds__(256) void cl_proj_kernel(
    const float* __restrict__ x1, const float* __restrict__ x2,
    const float* __restrict__ W1, const float* __restrict__ b1,
    const float* __restrict__ W2, const float* __restrict__ b2,
    _Float16* __restrict__ z1h, _Float16* __restrict__ z2h)
{
    int t = blockIdx.x * 256 + threadIdx.x;          // [0, 2N)
    const float* x  = (t < NROWS) ? x1 : x2;
    _Float16*   out = (t < NROWS) ? z1h : z2h;
    int row = t & (NROWS - 1);

    float xr[HID], h[HID];
    const float4* xp = (const float4*)(x + (size_t)row * HID);
#pragma unroll
    for (int q = 0; q < HID / 4; q++) {
        float4 v = xp[q];
        xr[4*q+0] = v.x; xr[4*q+1] = v.y; xr[4*q+2] = v.z; xr[4*q+3] = v.w;
    }
    for (int j = 0; j < HID; j++) {
        float acc = b1[j];
        const float* w = W1 + j * HID;
#pragma unroll
        for (int k = 0; k < HID; k++) acc = fmaf(xr[k], w[k], acc);
        h[j] = (acc > 0.0f) ? acc : expm1f(acc);     // ELU, alpha=1
    }
    float ss = 0.0f;
    for (int j = 0; j < HID; j++) {
        float acc = b2[j];
        const float* w = W2 + j * HID;
#pragma unroll
        for (int k = 0; k < HID; k++) acc = fmaf(h[k], w[k], acc);
        xr[j] = acc; ss += acc * acc;
    }
    float scale = 1.0f / fmaxf(sqrtf(ss), 1e-12f);   // F.normalize semantics
#pragma unroll
    for (int k = 0; k < HID; k++)
        out[(size_t)row * HID + k] = (_Float16)(xr[k] * scale);
}

// ---------------------------------------------------------------------------
// 3) tile kernel: S = exp(z1n @ z2n^T / tau) on 128x128 tiles via WMMA.
//    Block-local LDS reduction -> plain stores of per-tile partials (no global
//    atomics). Partial layout: rowSumP[bj][row], rowPosP[bj][row],
//    colSumP[bi][col], mpPosP[bi][col]  (each [64][8192] f32)
// ---------------------------------------------------------------------------
#define LDS_STRIDE 72   // halves per row: conflict-free b128 LDS reads, 16B aligned

__global__ __launch_bounds__(256) void cl_tile_kernel(
    const _Float16* __restrict__ z1h, const _Float16* __restrict__ z2h,
    const unsigned* __restrict__ Pb,
    float* __restrict__ rowSumP, float* __restrict__ rowPosP,
    float* __restrict__ colSumP, float* __restrict__ mpPosP)
{
    __shared__ _Float16 As[128 * LDS_STRIDE];
    __shared__ _Float16 Bs[128 * LDS_STRIDE];
    __shared__ float lRowS[128], lRowP[128], lColS[128], lColP[128];

    const int tid = threadIdx.x;
    const int rowBase = blockIdx.y * 128;
    const int colBase = blockIdx.x * 128;

    if (tid < 128) { lRowS[tid] = 0.f; lRowP[tid] = 0.f; lColS[tid] = 0.f; lColP[tid] = 0.f; }

    // stage z tiles into LDS (128 rows x 64 halves, padded stride) — uniform loop
    for (int c = tid; c < 1024; c += 256) {        // 1024 chunks of 8 halves
        int r = c >> 3, cc = (c & 7) * 8;
        *(v8h*)&As[r * LDS_STRIDE + cc] = *(const v8h*)&z1h[(size_t)(rowBase + r) * HID + cc];
        *(v8h*)&Bs[r * LDS_STRIDE + cc] = *(const v8h*)&z2h[(size_t)(colBase + r) * HID + cc];
    }
    __syncthreads();

    const int wave = tid >> 5, lane = tid & 31, lr = lane & 15;
    const bool hihalf = (lane >= 16);
    const int waveM = (wave & 3) * 32;
    const int waveN = (wave >> 2) * 64;
    const int aoff = hihalf ? 8 : 0;    // A: K chunks {aoff..aoff+7, aoff+16..aoff+23}
    const int boff = hihalf ? 8 : 0;    // B: K chunks {boff..boff+7, boff+16..boff+23}

    v8f acc[2][4];
#pragma unroll
    for (int mi = 0; mi < 2; mi++)
#pragma unroll
        for (int ni = 0; ni < 4; ni++)
#pragma unroll
            for (int r = 0; r < 8; r++) acc[mi][ni][r] = 0.0f;

#pragma unroll
    for (int k0 = 0; k0 < HID; k0 += 32) {
        v16h a[2], b[4];
#pragma unroll
        for (int mi = 0; mi < 2; mi++) {
            const _Float16* p = &As[(waveM + mi * 16 + lr) * LDS_STRIDE + k0 + aoff];
            v8h lo = *(const v8h*)p;
            v8h hi = *(const v8h*)(p + 16);
            a[mi] = __builtin_shufflevector(lo, hi, 0,1,2,3,4,5,6,7,8,9,10,11,12,13,14,15);
        }
#pragma unroll
        for (int ni = 0; ni < 4; ni++) {
            const _Float16* p = &Bs[(waveN + ni * 16 + lr) * LDS_STRIDE + k0 + boff];
            v8h lo = *(const v8h*)p;
            v8h hi = *(const v8h*)(p + 16);
            b[ni] = __builtin_shufflevector(lo, hi, 0,1,2,3,4,5,6,7,8,9,10,11,12,13,14,15);
        }
#pragma unroll
        for (int mi = 0; mi < 2; mi++)
#pragma unroll
            for (int ni = 0; ni < 4; ni++)
                acc[mi][ni] = __builtin_amdgcn_wmma_f32_16x16x32_f16(
                    false, a[mi], false, b[ni], (short)0, acc[mi][ni], false, false);
    }

    // ---- epilogue: s = exp(dot/tau); reduce 4 statistics into LDS ----
    const int wBase = (colBase + waveN) >> 5;        // even -> u64-aligned word pair
    float cs[4] = {0.f, 0.f, 0.f, 0.f};              // per-ni column sums (over 32 rows)
    float mp[4] = {0.f, 0.f, 0.f, 0.f};

#pragma unroll
    for (int mi = 0; mi < 2; mi++) {
        float rs[8], rp[8];
#pragma unroll
        for (int r = 0; r < 8; r++) { rs[r] = 0.0f; rp[r] = 0.0f; }
        const int gmBase = rowBase + waveM + mi * 16 + (hihalf ? 8 : 0);
        const int gmWord = gmBase >> 5;
        const int gmBit  = gmBase & 31;   // in {0,8,16,24}; +r stays < 32
        unsigned w2v[4];
#pragma unroll
        for (int ni = 0; ni < 4; ni++) {
            const int gn = colBase + waveN + ni * 16 + lr;
            w2v[ni] = Pb[(size_t)gn * (NROWS / 32) + gmWord];   // P[gn, gmBase..+31]
        }
#pragma unroll
        for (int r = 0; r < 8; r++) {
            const int gm = gmBase + r;
            // one u64 covers P[gm, colBase+waveN .. +63] -> all 4 ni bits
            const unsigned long long w1 =
                *(const unsigned long long*)&Pb[(size_t)gm * (NROWS / 32) + wBase];
#pragma unroll
            for (int ni = 0; ni < 4; ni++) {
                float s  = __expf(acc[mi][ni][r] * INV_TAU);
                float p1 = (float)((unsigned)(w1 >> (ni * 16 + lr)) & 1u);
                float p2 = (float)((w2v[ni] >> (gmBit + r)) & 1u);
                rs[r]  += s;
                rp[r]  += s * p1;
                cs[ni] += s;
                mp[ni] += s * p2;
            }
        }
        // rows: butterfly over the 16-lane group (sums the 16 columns x 4 ni)
#pragma unroll
        for (int m = 1; m < 16; m <<= 1) {
#pragma unroll
            for (int r = 0; r < 8; r++) {
                rs[r] += __shfl_xor(rs[r], m, 32);
                rp[r] += __shfl_xor(rp[r], m, 32);
            }
        }
        if (lr == 0) {
            const int lrow = waveM + mi * 16 + (hihalf ? 8 : 0);
#pragma unroll
            for (int r = 0; r < 8; r++) {
                atomicAdd(&lRowS[lrow + r], rs[r]);   // ds_add_f32
                atomicAdd(&lRowP[lrow + r], rp[r]);
            }
        }
    }
    // columns: lane l and l+16 share a column -> combine, 16 lanes add
#pragma unroll
    for (int ni = 0; ni < 4; ni++) {
        float c = cs[ni] + __shfl_xor(cs[ni], 16, 32);
        float m = mp[ni] + __shfl_xor(mp[ni], 16, 32);
        if (!hihalf) {
            const int lcol = waveN + ni * 16 + lr;
            atomicAdd(&lColS[lcol], c);
            atomicAdd(&lColP[lcol], m);
        }
    }
    __syncthreads();

    // plain coalesced stores of per-tile partials (no global atomics)
    if (tid < 128) {
        rowSumP[(size_t)blockIdx.x * NROWS + rowBase + tid] = lRowS[tid];
        rowPosP[(size_t)blockIdx.x * NROWS + rowBase + tid] = lRowP[tid];
        colSumP[(size_t)blockIdx.y * NROWS + colBase + tid] = lColS[tid];
        mpPosP [(size_t)blockIdx.y * NROWS + colBase + tid] = lColP[tid];
    }
}

// ---------------------------------------------------------------------------
// 4) fold the 64 per-tile partials per row, compute per-row loss, block partial
// ---------------------------------------------------------------------------
__global__ __launch_bounds__(256) void cl_reduce_kernel(
    const float* __restrict__ rowSumP, const float* __restrict__ rowPosP,
    const float* __restrict__ colSumP, const float* __restrict__ mpPosP,
    float* __restrict__ blockPart)
{
    int i = blockIdx.x * 256 + threadIdx.x;     // row index [0, 8192)
    float rs = 0.f, rp = 0.f, cs = 0.f, mp = 0.f;
    for (int t = 0; t < NTILES; t++) {
        rs += rowSumP[(size_t)t * NROWS + i];
        rp += rowPosP[(size_t)t * NROWS + i];
        cs += colSumP[(size_t)t * NROWS + i];
        mp += mpPosP [(size_t)t * NROWS + i];
    }
    float sc = -logf(rp / (rs + 1e-8f) + 1e-8f);
    float ml = -logf(mp / (cs + 1e-8f) + 1e-8f);
    float local = LAMBDA * sc + (1.0f - LAMBDA) * ml;

    __shared__ float red[256];
    red[threadIdx.x] = local;
    __syncthreads();
    for (int s = 128; s > 0; s >>= 1) {
        if (threadIdx.x < s) red[threadIdx.x] += red[threadIdx.x + s];
        __syncthreads();
    }
    if (threadIdx.x == 0) blockPart[blockIdx.x] = red[0];
}

// 5) deterministic finish: one wave sums the 32 block partials
__global__ void cl_final_kernel(const float* __restrict__ blockPart, float* __restrict__ out) {
    float v = blockPart[threadIdx.x];          // 32 threads
#pragma unroll
    for (int m = 16; m > 0; m >>= 1) v += __shfl_xor(v, m, 32);
    if (threadIdx.x == 0) out[0] = v / (float)NROWS;
}

// ---------------------------------------------------------------------------
extern "C" void kernel_launch(void* const* d_in, const int* in_sizes, int n_in,
                              void* d_out, int out_size, void* d_ws, size_t ws_size,
                              hipStream_t stream) {
    const float* x1 = (const float*)d_in[0];
    const float* x2 = (const float*)d_in[1];
    const float* W1 = (const float*)d_in[2];
    const float* b1 = (const float*)d_in[3];
    const float* W2 = (const float*)d_in[4];
    const float* b2 = (const float*)d_in[5];
    const float* P  = (const float*)d_in[6];
    float* out = (float*)d_out;

    // workspace layout (bytes):
    //   [ 0, 1MB)   z1h  f16 [8192][64]
    //   [ 1, 2MB)   z2h  f16 [8192][64]
    //   [ 2,10MB)   Pb   u32 bitmask [8192][256]
    //   [10,18MB)   partials: rowSumP/rowPosP/colSumP/mpPosP, each [64][8192] f32
    //   [18MB..)    blockPart f32 [32]
    char* ws = (char*)d_ws;
    _Float16* z1h  = (_Float16*)(ws);
    _Float16* z2h  = (_Float16*)(ws + (1u << 20));
    unsigned* Pb   = (unsigned*)(ws + (2u << 20));
    float* rowSumP = (float*)(ws + (10u << 20));
    float* rowPosP = rowSumP + (size_t)NTILES * NROWS;
    float* colSumP = rowPosP + (size_t)NTILES * NROWS;
    float* mpPosP  = colSumP + (size_t)NTILES * NROWS;
    float* blockPart = mpPosP + (size_t)NTILES * NROWS;

    cl_pack_kernel<<<(NROWS * (NROWS / 256)), 256, 0, stream>>>(P, Pb);   // 262144 blocks
    cl_proj_kernel<<<(2 * NROWS) / 256, 256, 0, stream>>>(x1, x2, W1, b1, W2, b2, z1h, z2h);
    cl_tile_kernel<<<dim3(NROWS / 128, NROWS / 128), 256, 0, stream>>>(
        z1h, z2h, Pb, rowSumP, rowPosP, colSumP, mpPosP);
    cl_reduce_kernel<<<NROWS / 256, 256, 0, stream>>>(rowSumP, rowPosP, colSumP, mpPosP, blockPart);
    cl_final_kernel<<<1, 32, 0, stream>>>(blockPart, out);
}
